// AtomAttentionPairBias_69097433858587
// MI455X (gfx1250) — compile-verified, matrix operand order checked by reading, not run on hardware
//
#include <hip/hip_runtime.h>

#define BSZ    2
#define NATOMS 8192
#define CA     128
#define LDC    136              // LDS row pitch: 128 + 8 bf16 pad (16B) -> bank-conflict-free frag loads
#define NH     8
#define DHD    16
#define NQW    32
#define NKW    128
#define CPAIR  16
#define NBW    (NATOMS / NQW)   // 256
#define PADW   48

typedef __attribute__((ext_vector_type(16))) __bf16 v16bf;
typedef __attribute__((ext_vector_type(8)))  float  v8f;

static __device__ inline v8f vzero8() {
  v8f z;
#pragma unroll
  for (int i = 0; i < 8; ++i) z[i] = 0.0f;
  return z;
}

static __device__ inline float sigm(float x) { return 1.0f / (1.0f + __expf(-x)); }

// Load a 16x32 bf16 fragment (A or B operand) from a row-major [rows][ld] LDS tile.
// Layout per ISA 7.12.2 (16-bit A 16x32): lanes 0-15 -> K {0..7,16..23}, lanes 16-31 -> K {8..15,24..31}.
static __device__ inline v16bf frag_ld(const __bf16* base, int row0, int col0, int ld) {
  const int lane = threadIdx.x & 31;
  const __bf16* p = base + (row0 + (lane & 15)) * ld + col0 + ((lane >> 4) << 3);
  v16bf f;
#pragma unroll
  for (int i = 0; i < 8; ++i) f[i] = p[i];
#pragma unroll
  for (int i = 0; i < 8; ++i) f[i + 8] = p[i + 16];
  return f;
}

// Same, but only K=0..15 valid (upper half of the K=32 fragment zeroed) — used for
// per-head DH=16 score contraction inside a K=32 WMMA.
static __device__ inline v16bf frag_ld_k16(const __bf16* base, int row0, int col0, int ld) {
  const int lane = threadIdx.x & 31;
  const __bf16* p = base + (row0 + (lane & 15)) * ld + col0 + ((lane >> 4) << 3);
  v16bf f;
#pragma unroll
  for (int i = 0; i < 8; ++i) f[i] = p[i];
#pragma unroll
  for (int i = 0; i < 8; ++i) f[i + 8] = (__bf16)0.0f;
  return f;
}

static __device__ inline v8f wmma_bf(v16bf a, v16bf b, v8f c) {
  return __builtin_amdgcn_wmma_f32_16x16x32_bf16(false, a, false, b, (short)0, c, false, false);
}

// ---------------------------------------------------------------------------
// Kernel 1: fused AdaLN + a = sigmoid(s_ln@Wg + bg)*a_ln + s_ln@Ws, then
// Q = 0.25*(a@wq), K = a@wk, V = a@wv, G = sigmoid(a@wg + bgate), all bf16.
// 64 atoms per workgroup, 8 waves, 6 WMMA GEMM passes through one LDS W^T tile.
// ---------------------------------------------------------------------------
__global__ __launch_bounds__(256, 1)
void k_prep(const float* __restrict__ arep, const float* __restrict__ sproj,
            const float* __restrict__ slnw,
            const float* __restrict__ gw, const float* __restrict__ gb,
            const float* __restrict__ skw,
            const float* __restrict__ wq, const float* __restrict__ wk,
            const float* __restrict__ wv, const float* __restrict__ wg,
            const float* __restrict__ wgb,
            __bf16* __restrict__ Qo, __bf16* __restrict__ Ko,
            __bf16* __restrict__ Vo, __bf16* __restrict__ Go)
{
  constexpr int ROWS = 64;
  __shared__ __bf16 sln[ROWS * LDC];   // ~17 KB
  __shared__ __bf16 aln[ROWS * LDC];   // ~17 KB
  __shared__ __bf16 wt[CA * LDC];      // ~35 KB, W^T [out][in]
  __shared__ __bf16 abuf[ROWS * LDC];  // ~17 KB

  const int tid  = threadIdx.x;
  const int wave = tid >> 5, lane = tid & 31;
  const int r0   = blockIdx.x * ROWS;  // flat row over BS*N_ATOMS

  // ---- LayerNorms: one row per wave pass, lane covers 4 channels ----
  {
    const float4 w4 = *(const float4*)(slnw + lane * 4);
    for (int j = 0; j < ROWS / 8; ++j) {
      const int row = wave * (ROWS / 8) + j;
      const float4 xa = *(const float4*)(arep  + (size_t)(r0 + row) * CA + lane * 4);
      const float4 xs = *(const float4*)(sproj + (size_t)(r0 + row) * CA + lane * 4);
      float sa = xa.x + xa.y + xa.z + xa.w;
      float qa = xa.x * xa.x + xa.y * xa.y + xa.z * xa.z + xa.w * xa.w;
      float ss = xs.x + xs.y + xs.z + xs.w;
      float qs = xs.x * xs.x + xs.y * xs.y + xs.z * xs.z + xs.w * xs.w;
#pragma unroll
      for (int m = 16; m; m >>= 1) {
        sa += __shfl_xor(sa, m); qa += __shfl_xor(qa, m);
        ss += __shfl_xor(ss, m); qs += __shfl_xor(qs, m);
      }
      const float mua = sa * (1.0f / CA), mus = ss * (1.0f / CA);
      const float ra = rsqrtf(qa * (1.0f / CA) - mua * mua + 1e-5f);
      const float rs = rsqrtf(qs * (1.0f / CA) - mus * mus + 1e-5f);
      __bf16* ap = aln + row * LDC + lane * 4;
      __bf16* sp = sln + row * LDC + lane * 4;
      ap[0] = (__bf16)((xa.x - mua) * ra); ap[1] = (__bf16)((xa.y - mua) * ra);
      ap[2] = (__bf16)((xa.z - mua) * ra); ap[3] = (__bf16)((xa.w - mua) * ra);
      sp[0] = (__bf16)((xs.x - mus) * rs * w4.x); sp[1] = (__bf16)((xs.y - mus) * rs * w4.y);
      sp[2] = (__bf16)((xs.z - mus) * rs * w4.z); sp[3] = (__bf16)((xs.w - mus) * rs * w4.w);
    }
  }
  __syncthreads();

  const int mt  = wave & 3;          // 4 m-tiles of 16 rows
  const int nb4 = (wave >> 2) << 2;  // n-tile group {0..3} or {4..7}

  auto loadwt = [&](const float* W) {
    for (int i = tid; i < CA * CA; i += 256)
      wt[(i & 127) * LDC + (i >> 7)] = (__bf16)W[i];   // transpose f32 -> bf16
  };
  auto gemm4 = [&](const __bf16* A, v8f acc[4]) {
#pragma unroll
    for (int kb = 0; kb < 4; ++kb) {
      const v16bf af = frag_ld(A, mt * 16, kb * 32, LDC);
#pragma unroll
      for (int no = 0; no < 4; ++no) {
        const v16bf bf = frag_ld(wt, (nb4 + no) * 16, kb * 32, LDC);
        acc[no] = wmma_bf(af, bf, acc[no]);
      }
    }
  };

  // ---- gate GEMM ----
  loadwt(gw);
  __syncthreads();
  v8f accg[4];
#pragma unroll
  for (int no = 0; no < 4; ++no) accg[no] = vzero8();
  gemm4(sln, accg);
  float gpart[4][8];
#pragma unroll
  for (int no = 0; no < 4; ++no)
#pragma unroll
    for (int i = 0; i < 8; ++i) {
      const int M  = mt * 16 + ((lane < 16) ? i : 8 + i);
      const int Nc = (nb4 + no) * 16 + (lane & 15);
      gpart[no][i] = sigm(accg[no][i] + gb[Nc]) * (float)aln[M * LDC + Nc];
    }
  __syncthreads();

  // ---- skip GEMM; a = gate*a_ln + skip ----
  loadwt(skw);
  __syncthreads();
  v8f accs[4];
#pragma unroll
  for (int no = 0; no < 4; ++no) accs[no] = vzero8();
  gemm4(sln, accs);
#pragma unroll
  for (int no = 0; no < 4; ++no)
#pragma unroll
    for (int i = 0; i < 8; ++i) {
      const int M  = mt * 16 + ((lane < 16) ? i : 8 + i);
      const int Nc = (nb4 + no) * 16 + (lane & 15);
      abuf[M * LDC + Nc] = (__bf16)(gpart[no][i] + accs[no][i]);
    }
  __syncthreads();

  // ---- Q/K/V/G GEMMs ----
  const float* Ws[4] = {wq, wk, wv, wg};
  __bf16*      Os[4] = {Qo, Ko, Vo, Go};
#pragma unroll
  for (int t = 0; t < 4; ++t) {
    loadwt(Ws[t]);
    __syncthreads();
    v8f acc[4];
#pragma unroll
    for (int no = 0; no < 4; ++no) acc[no] = vzero8();
    gemm4(abuf, acc);
#pragma unroll
    for (int no = 0; no < 4; ++no)
#pragma unroll
      for (int i = 0; i < 8; ++i) {
        const int M  = mt * 16 + ((lane < 16) ? i : 8 + i);
        const int Nc = (nb4 + no) * 16 + (lane & 15);
        float x = acc[no][i];
        if (t == 0)      x *= 0.25f;                 // q scale = 1/sqrt(DH)
        else if (t == 3) x = sigm(x + wgb[Nc]);      // gate
        Os[t][(size_t)(r0 + M) * CA + Nc] = (__bf16)x;
      }
    __syncthreads();
  }
}

// ---------------------------------------------------------------------------
// Kernel 2: one workgroup per (batch, window). One wave per head.
// prefetch pair strip -> scores -> fused pair-bias LN + mask -> softmax ->
// P@V -> gate -> wo -> out_w
// ---------------------------------------------------------------------------
__global__ __launch_bounds__(256, 1)
void k_attn(const __bf16* __restrict__ Qg, const __bf16* __restrict__ Kg,
            const __bf16* __restrict__ Vg, const __bf16* __restrict__ Gg,
            const float* __restrict__ pl, const float* __restrict__ mask,
            const float* __restrict__ plw, const float* __restrict__ plb,
            const float* __restrict__ pw,
            const float* __restrict__ wo, const float* __restrict__ wob,
            const float* __restrict__ ow, const float* __restrict__ ob,
            float* __restrict__ out)
{
  // Q + K + V^T + per-wave P strips + Ob, all row-padded to LDC; ~122 KB total
  __shared__ __align__(16) char smem[(NQW*LDC + NKW*LDC + NKW*LDC + 8*16*LDC + NQW*LDC) * 2 + CPAIR*NH*4];
  __bf16* Qs  = (__bf16*)smem;
  __bf16* Ks  = Qs  + NQW * LDC;
  __bf16* VTs = Ks  + NKW * LDC;     // [d][k]
  __bf16* Ps  = VTs + NKW * LDC;     // 8 waves x [16][LDC]
  __bf16* Ob  = Ps  + 8 * 16 * LDC;  // gated attention output [32][128]
  float*  pws = (float*)(Ob + NQW * LDC);

  const int tid  = threadIdx.x, wave = tid >> 5, lane = tid & 31;
  const int b    = blockIdx.x >> 8;
  const int w    = blockIdx.x & (NBW - 1);
  const int qa0  = b * NATOMS + w * NQW;   // first query row (flat)
  const int ka0  = w * NQW - PADW;         // in-batch atom index of key 0

  for (int i = tid; i < NQW * CA; i += 256)
    Qs[(i >> 7) * LDC + (i & 127)] = Qg[(size_t)qa0 * CA + i];
  for (int i = tid; i < NKW * CA; i += 256) {
    const int kk = i >> 7, c = i & 127;
    const int ka = ka0 + kk;
    const bool v = (ka >= 0) && (ka < NATOMS);
    const size_t gi = (size_t)(b * NATOMS + (v ? ka : 0)) * CA + c;
    Ks[kk * LDC + c]  = v ? Kg[gi] : (__bf16)0.0f;
    VTs[c * LDC + kk] = v ? Vg[gi] : (__bf16)0.0f;
  }
  if (tid < CPAIR * NH) pws[tid] = pw[tid];
  __syncthreads();

  float lw[16], lb[16];
#pragma unroll
  for (int c = 0; c < 16; c += 4) {
    const float4 t = *(const float4*)(plw + c);
    const float4 u = *(const float4*)(plb + c);
    lw[c] = t.x; lw[c+1] = t.y; lw[c+2] = t.z; lw[c+3] = t.w;
    lb[c] = u.x; lb[c+1] = u.y; lb[c+2] = u.z; lb[c+3] = u.w;
  }

  const int h = wave;                 // 8 waves == 8 heads
  __bf16* Pw = Ps + wave * 16 * LDC;  // per-wave probability strip
  const size_t plbase = (size_t)(b * NBW + w) * NQW * NKW * CPAIR;

#pragma unroll 1
  for (int m = 0; m < 2; ++m) {       // 2 query tiles of 16
    // ---- prefetch this strip's pair-local cachelines (the dominant HBM
    //      stream: 2 adjacent kk positions share a 128B line -> even lanes) ----
    if ((lane & 1) == 0) {
#pragma unroll
      for (int n = 0; n < 8; ++n)
#pragma unroll
        for (int i = 0; i < 8; ++i) {
          const int q  = m * 16 + ((lane < 16) ? i : 8 + i);
          const int kk = n * 16 + (lane & 15);
          __builtin_prefetch(pl + plbase + ((size_t)q * NKW + kk) * CPAIR, 0, 0);
        }
    }

    v8f acc[8];
#pragma unroll
    for (int n = 0; n < 8; ++n) acc[n] = vzero8();

    // ---- scores: Q[16x16] . K^T, DH=16 packed into low half of K=32 WMMA ----
    const v16bf aq = frag_ld_k16(Qs, m * 16, h * DHD, LDC);
#pragma unroll
    for (int n = 0; n < 8; ++n) {
      const v16bf bk = frag_ld_k16(Ks, n * 16, h * DHD, LDC);
      acc[n] = wmma_bf(aq, bk, acc[n]);
    }

    // ---- fused pair-bias (LN16 -> @pair_w[:,h]) + attention mask, at D-layout ----
#pragma unroll
    for (int n = 0; n < 8; ++n)
#pragma unroll
      for (int i = 0; i < 8; ++i) {
        const int q  = m * 16 + ((lane < 16) ? i : 8 + i);
        const int kk = n * 16 + (lane & 15);
        const float* pp = pl + plbase + ((size_t)q * NKW + kk) * CPAIR;
        float x[16];
#pragma unroll
        for (int c = 0; c < 16; c += 4) {
          const float4 t = *(const float4*)(pp + c);
          x[c] = t.x; x[c+1] = t.y; x[c+2] = t.z; x[c+3] = t.w;
        }
        float s = 0.0f, q2 = 0.0f;
#pragma unroll
        for (int c = 0; c < 16; ++c) { s += x[c]; q2 += x[c] * x[c]; }
        const float mu = s * (1.0f / 16.0f);
        const float r  = rsqrtf(q2 * (1.0f / 16.0f) - mu * mu + 1e-5f);
        float bias = 0.0f;
#pragma unroll
        for (int c = 0; c < 16; ++c)
          bias += ((x[c] - mu) * r * lw[c] + lb[c]) * pws[c * NH + h];
        const int ka = ka0 + kk;
        float mb;
        if (ka >= 0 && ka < NATOMS)
          mb = -1.0e6f * (1.0f - mask[qa0 + q] * mask[b * NATOMS + ka]);
        else
          mb = -1.0e8f;
        acc[n][i] += bias + mb;
      }

    // ---- softmax over NK=128: 8 local vals x 16-lane half-wave reduction ----
#pragma unroll
    for (int i = 0; i < 8; ++i) {
      float mx = acc[0][i];
#pragma unroll
      for (int n = 1; n < 8; ++n) mx = fmaxf(mx, acc[n][i]);
#pragma unroll
      for (int msk = 8; msk; msk >>= 1) mx = fmaxf(mx, __shfl_xor(mx, msk));
      float sum = 0.0f;
#pragma unroll
      for (int n = 0; n < 8; ++n) { const float e = __expf(acc[n][i] - mx); acc[n][i] = e; sum += e; }
#pragma unroll
      for (int msk = 8; msk; msk >>= 1) sum += __shfl_xor(sum, msk);
      const float inv = 1.0f / sum;
#pragma unroll
      for (int n = 0; n < 8; ++n) acc[n][i] *= inv;
    }

    // ---- stage P to LDS (D-layout -> row-major), then O = P @ V ----
#pragma unroll
    for (int n = 0; n < 8; ++n)
#pragma unroll
      for (int i = 0; i < 8; ++i) {
        const int rl = (lane < 16) ? i : 8 + i;
        Pw[rl * LDC + n * 16 + (lane & 15)] = (__bf16)acc[n][i];
      }
    v8f accO = vzero8();
#pragma unroll
    for (int kb = 0; kb < 4; ++kb) {
      const v16bf ap = frag_ld(Pw, 0, kb * 32, LDC);
      const v16bf bv = frag_ld(VTs, h * DHD, kb * 32, LDC);
      accO = wmma_bf(ap, bv, accO);
    }
    // ---- apply sigmoid gate and assemble Ob[32][128] ----
#pragma unroll
    for (int i = 0; i < 8; ++i) {
      const int q = m * 16 + ((lane < 16) ? i : 8 + i);
      const int d = lane & 15;
      const float g = (float)Gg[(size_t)(qa0 + q) * CA + h * DHD + d];
      Ob[q * LDC + h * DHD + d] = (__bf16)(accO[i] * g);
    }
  }
  __syncthreads();

  // ---- o @ wo + wo_b ----  (reuse Ks as W^T, VTs as f32 oproj, Qs as bf16 oproj)
  __bf16* WT   = Ks;
  float*  opf  = (float*)VTs;
  __bf16* OP16 = Qs;
  for (int i = tid; i < CA * CA; i += 256) WT[(i & 127) * LDC + (i >> 7)] = (__bf16)wo[i];
  __syncthreads();
  {
    const int n = wave;
#pragma unroll 1
    for (int m = 0; m < 2; ++m) {
      v8f acc = vzero8();
#pragma unroll
      for (int kb = 0; kb < 4; ++kb)
        acc = wmma_bf(frag_ld(Ob, m * 16, kb * 32, LDC), frag_ld(WT, n * 16, kb * 32, LDC), acc);
#pragma unroll
      for (int i = 0; i < 8; ++i) {
        const int q   = m * 16 + ((lane < 16) ? i : 8 + i);
        const int col = n * 16 + (lane & 15);
        const float x = acc[i] + wob[col];
        opf[q * LDC + col]  = x;
        OP16[q * LDC + col] = (__bf16)x;
      }
    }
  }
  __syncthreads();

  // ---- out = sigmoid(o @ out_w + out_b) * o ----
  for (int i = tid; i < CA * CA; i += 256) WT[(i & 127) * LDC + (i >> 7)] = (__bf16)ow[i];
  __syncthreads();
  {
    const int n = wave;
#pragma unroll 1
    for (int m = 0; m < 2; ++m) {
      v8f acc = vzero8();
#pragma unroll
      for (int kb = 0; kb < 4; ++kb)
        acc = wmma_bf(frag_ld(OP16, m * 16, kb * 32, LDC), frag_ld(WT, n * 16, kb * 32, LDC), acc);
#pragma unroll
      for (int i = 0; i < 8; ++i) {
        const int q   = m * 16 + ((lane < 16) ? i : 8 + i);
        const int col = n * 16 + (lane & 15);
        const float x = acc[i] + ob[col];
        out[(size_t)(qa0 + q) * CA + col] = sigm(x) * opf[q * LDC + col];
      }
    }
  }
}

extern "C" void kernel_launch(void* const* d_in, const int* in_sizes, int n_in,
                              void* d_out, int out_size, void* d_ws, size_t ws_size,
                              hipStream_t stream) {
  const float* arep   = (const float*)d_in[0];
  const float* sproj  = (const float*)d_in[1];
  const float* pl     = (const float*)d_in[2];
  const float* mask   = (const float*)d_in[3];
  const float* slnw   = (const float*)d_in[4];
  const float* gw     = (const float*)d_in[5];
  const float* gb     = (const float*)d_in[6];
  const float* skw    = (const float*)d_in[7];
  const float* plw    = (const float*)d_in[8];
  const float* plb    = (const float*)d_in[9];
  const float* pw     = (const float*)d_in[10];
  const float* wq     = (const float*)d_in[11];
  const float* wk     = (const float*)d_in[12];
  const float* wv     = (const float*)d_in[13];
  const float* wg     = (const float*)d_in[14];
  const float* wgb    = (const float*)d_in[15];
  const float* wo     = (const float*)d_in[16];
  const float* wob    = (const float*)d_in[17];
  const float* ow     = (const float*)d_in[18];
  const float* ob     = (const float*)d_in[19];

  const size_t nrows = (size_t)BSZ * NATOMS;           // 16384
  __bf16* Q = (__bf16*)d_ws;                            // 4 MB each, 16 MB total
  __bf16* K = Q + nrows * CA;
  __bf16* V = K + nrows * CA;
  __bf16* G = V + nrows * CA;

  k_prep<<<(int)(nrows / 64), 256, 0, stream>>>(arep, sproj, slnw, gw, gb, skw,
                                                wq, wk, wv, wg, wgb, Q, K, V, G);
  k_attn<<<BSZ * NBW, 256, 0, stream>>>(Q, K, V, G, pl, mask, plw, plb, pw,
                                        wo, wob, ow, ob, (float*)d_out);
}
